// EMAPath_14748917694570
// MI455X (gfx1250) — matrix operand order repaired and checked by reading, not compiled, not run on hardware
//
#include <hip/hip_runtime.h>
#include <cmath>

// Problem constants (match reference)
constexpr int BB  = 16;
constexpr int TT  = 8192;
constexpr int DIN = 64;
constexpr int DM  = 128;

constexpr int CHUNK = 64;           // timesteps per block
constexpr int NC    = TT / CHUNK;   // 128 chunks along T
constexpr int NTHR  = 256;          // 8 waves (wave32)

typedef __attribute__((ext_vector_type(16))) __bf16 v16bf;
typedef __attribute__((ext_vector_type(8)))  float  v8f;
typedef __attribute__((ext_vector_type(4)))  unsigned int u32x4;
typedef __attribute__((ext_vector_type(8)))  int    i32x8;
typedef __attribute__((ext_vector_type(4)))  int    i32x4;

static __device__ __forceinline__ __bf16 f2bf(float f) {
  return (__bf16)f;   // native v_cvt on gfx1250 (RTNE)
}

static __device__ __forceinline__ float fast_sigmoid(float v) {
  // 1/(1+exp(-v)) with v_exp_f32 + v_rcp_f32 (no IEEE div sequence)
  return __builtin_amdgcn_rcpf(1.0f + __expf(-v));
}

// --- Tensor Data Mover: 2D f32 tile (global row-major) -> LDS (packed rows) ---
// D# per CDNA5 ISA ch.8: group0 = {count/flags, lds_addr, global_addr, type=2},
// group1 = {data_size=4B, tensor dims, tile dims, dim0 stride}. Groups 2/3 zero (2D).
static __device__ __forceinline__ void tdm_load_2d_f32(unsigned lds_byte_addr,
                                                       const void* gptr,
                                                       unsigned tile_d0,   // elems per row
                                                       unsigned tile_d1,   // rows
                                                       unsigned stride0)   // row stride (elems)
{
  unsigned long long ga = (unsigned long long)(uintptr_t)gptr;
  u32x4 g0;
  g0[0] = 1u;                                            // count=1, user mode
  g0[1] = lds_byte_addr;                                 // lds_addr [63:32]
  g0[2] = (unsigned)(ga & 0xFFFFFFFFu);                  // global_addr lo
  g0[3] = (unsigned)((ga >> 32) & 0x01FFFFFFu) | (2u << 30); // addr[56:32] | type=2

  i32x8 g1;
  g1[0] = (int)(2u << 16);                               // workgroup_mask=0, data_size=2 (4B)
  g1[1] = (int)((tile_d0 & 0xFFFFu) << 16);              // tensor_dim0 lo16 -> [63:48]
  g1[2] = (int)(((tile_d0 >> 16) & 0xFFFFu) | ((tile_d1 & 0xFFFFu) << 16)); // dim0 hi | dim1 lo
  g1[3] = (int)(((tile_d1 >> 16) & 0xFFFFu) | ((tile_d0 & 0xFFFFu) << 16)); // dim1 hi | tile_dim0
  g1[4] = (int)(tile_d1 & 0xFFFFu);                      // tile_dim1 | tile_dim2=0
  g1[5] = (int)stride0;                                  // tensor_dim0_stride lo32
  g1[6] = 0;
  g1[7] = 0;

  i32x4 z4 = {0, 0, 0, 0};
#if __clang_major__ >= 23
  i32x8 z8 = {0, 0, 0, 0, 0, 0, 0, 0};
  __builtin_amdgcn_tensor_load_to_lds(g0, g1, z4, z4, z8, 0);
#else
  __builtin_amdgcn_tensor_load_to_lds(g0, g1, z4, z4, 0);
#endif
}

// Shared body for pass1 (aggregates) and pass3 (final y with carry).
// Block: 256 threads = 8 waves. Waves 0-3 compute proj (z), waves 4-7 compute gate (g);
// wave w owns two 16-column tiles starting at n0 = (w&3)*32.
template <bool FINAL>
static __device__ __forceinline__ void ema_body(const float* __restrict__ x,
                                                const float* __restrict__ pw,
                                                const float* __restrict__ pb,
                                                const float* __restrict__ gw,
                                                const float* __restrict__ gb,
                                                float* __restrict__ aggA,
                                                float* __restrict__ aggB,
                                                const float* __restrict__ carry,
                                                float* __restrict__ out)
{
  __shared__ float xbuf[CHUNK * DIN];       // 16 KB: x tile (f32, TDM-loaded)
  __shared__ float zg[2 * CHUNK * DM];      // 64 KB: z then g, [CHUNK x DM] each

  const int b    = blockIdx.y;
  const int ch   = blockIdx.x;
  const int t0   = ch * CHUNK;
  const int tid  = threadIdx.x;
  const int lane = tid & 31;
  const int h    = lane >> 4;               // half-wave (ISA fragment layouts)
  const int nl   = lane & 15;

  // Wave index as a proven-uniform (SGPR) value -> scalar branches, no EXEC juggling.
  const int wave   = __builtin_amdgcn_readfirstlane(tid >> 5);
  const bool isGate = wave >= 4;
  const int n0     = (wave & 3) * 32;

  // ---- async tensor load of the x tile into LDS (one wave issues) ----
  if (wave == 0) {
    tdm_load_2d_f32((unsigned)(uintptr_t)xbuf,
                    x + ((size_t)b * TT + t0) * DIN,
                    DIN, CHUNK, DIN);
    __builtin_amdgcn_s_wait_tensorcnt(0);
  }
  __syncthreads();

  const float* W    = isGate ? gw : pw;
  const float* bias = isGate ? gb : pb;

  // B fragments (loop-invariant): 32x16 bf16 per k-step.
  // Layout: lanes 0-15 -> N=lane, K=e; lanes 16-31 -> N=lane-16, K=16+e.
  v16bf bfrag[2][2];
#pragma unroll
  for (int nt = 0; nt < 2; ++nt) {
    const int col = n0 + nt * 16 + nl;
#pragma unroll
    for (int ks = 0; ks < 2; ++ks) {
#pragma unroll
      for (int e = 0; e < 16; ++e) {
        const int k = h * 16 + e + ks * 32;
        bfrag[nt][ks][e] = f2bf(W[k * DM + col]);
      }
    }
  }
  const float biasv[2] = { bias[n0 + nl], bias[n0 + 16 + nl] };

  float* const dst = zg + (isGate ? CHUNK * DM : 0);   // scalar select

  // ---- GEMM: CHUNK rows in strips of 16, K=64 as two 16x16x32 WMMAs ----
#pragma unroll
  for (int s = 0; s < CHUNK / 16; ++s) {
    // A fragment 16x32 bf16: lane row = nl; K runs {16h..16h+15} (+32 for ks=1)
    v16bf afrag[2];
#pragma unroll
    for (int ks = 0; ks < 2; ++ks) {
#pragma unroll
      for (int e = 0; e < 16; ++e) {
        const int k = h * 16 + e + ks * 32;
        afrag[ks][e] = f2bf(xbuf[(s * 16 + nl) * DIN + k]);
      }
    }
#pragma unroll
    for (int nt = 0; nt < 2; ++nt) {
      v8f acc;
#pragma unroll
      for (int i = 0; i < 8; ++i) acc[i] = biasv[nt];   // fold bias into C
      acc = __builtin_amdgcn_wmma_f32_16x16x32_bf16(false, afrag[0], false, bfrag[nt][0],
                                                    (short)0, acc, false, false);
      acc = __builtin_amdgcn_wmma_f32_16x16x32_bf16(false, afrag[1], false, bfrag[nt][1],
                                                    (short)0, acc, false, false);
      if (isGate) {                                     // scalar (wave-uniform) branch
#pragma unroll
        for (int i = 0; i < 8; ++i) acc[i] = fast_sigmoid(acc[i]);
      }
#pragma unroll
      for (int i = 0; i < 8; ++i) {
        // C layout: VGPR i -> row i + 8h, col nl (per half-wave)
        dst[(s * 16 + i + h * 8) * DM + n0 + nt * 16 + nl] = acc[i];
      }
    }
  }
  __syncthreads();

  // ---- serial scan over the chunk: one thread per channel ----
  if (tid < DM) {
    const int c = tid;
    const float* zb   = zg;
    const float* gbuf = zg + CHUNK * DM;
    if (FINAL) {
      float y = carry[((size_t)b * NC + ch) * DM + c];
      for (int t = 0; t < CHUNK; ++t) {
        const float g = gbuf[t * DM + c];
        const float z = zb[t * DM + c];
        y = fmaf(g, z - y, y);                           // y = g*z + (1-g)*y
        out[((size_t)b * TT + t0 + t) * DM + c] = y;
      }
    } else {
      float A = 1.0f, Bv = 0.0f;
      for (int t = 0; t < CHUNK; ++t) {
        const float g = gbuf[t * DM + c];
        const float z = zb[t * DM + c];
        const float a = 1.0f - g;
        Bv = fmaf(a, Bv, g * z);
        A *= a;
      }
      aggA[((size_t)b * NC + ch) * DM + c] = A;
      aggB[((size_t)b * NC + ch) * DM + c] = Bv;
    }
  }
}

__global__ __launch_bounds__(NTHR) void ema_pass1(const float* __restrict__ x,
                                                  const float* __restrict__ pw,
                                                  const float* __restrict__ pb,
                                                  const float* __restrict__ gw,
                                                  const float* __restrict__ gb,
                                                  float* __restrict__ aggA,
                                                  float* __restrict__ aggB)
{
  ema_body<false>(x, pw, pb, gw, gb, aggA, aggB, nullptr, nullptr);
}

__global__ __launch_bounds__(256) void ema_carry(const float* __restrict__ aggA,
                                                 const float* __restrict__ aggB,
                                                 float* __restrict__ carry)
{
  const int idx = blockIdx.x * blockDim.x + threadIdx.x;
  if (idx >= BB * DM) return;
  const int b = idx / DM;
  const int c = idx % DM;
  float run = 0.0f;  // y_{-1} = 0
  for (int j = 0; j < NC; ++j) {
    const size_t o = ((size_t)b * NC + j) * DM + c;
    carry[o] = run;
    run = fmaf(aggA[o], run, aggB[o]);
  }
}

__global__ __launch_bounds__(NTHR) void ema_pass3(const float* __restrict__ x,
                                                  const float* __restrict__ pw,
                                                  const float* __restrict__ pb,
                                                  const float* __restrict__ gw,
                                                  const float* __restrict__ gb,
                                                  const float* __restrict__ carry,
                                                  float* __restrict__ out)
{
  ema_body<true>(x, pw, pb, gw, gb, nullptr, nullptr, carry, out);
}

extern "C" void kernel_launch(void* const* d_in, const int* in_sizes, int n_in,
                              void* d_out, int out_size, void* d_ws, size_t ws_size,
                              hipStream_t stream)
{
  const float* x  = (const float*)d_in[0];
  const float* pw = (const float*)d_in[1];
  const float* pb = (const float*)d_in[2];
  const float* gw = (const float*)d_in[3];
  const float* gb = (const float*)d_in[4];
  float* out = (float*)d_out;

  float* aggA  = (float*)d_ws;
  float* aggB  = aggA + (size_t)BB * NC * DM;
  float* carry = aggB + (size_t)BB * NC * DM;

  dim3 grid(NC, BB), block(NTHR);
  ema_pass1<<<grid, block, 0, stream>>>(x, pw, pb, gw, gb, aggA, aggB);
  ema_carry<<<(BB * DM + 255) / 256, 256, 0, stream>>>(aggA, aggB, carry);
  ema_pass3<<<grid, block, 0, stream>>>(x, pw, pb, gw, gb, carry, out);
}
